// GatNet_33612414059062
// MI455X (gfx1250) — compile-verified, matrix operand order checked by reading, not run on hardware
//
#include <hip/hip_runtime.h>
#include <hip/hip_bf16.h>

// ---------------------------------------------------------------------------
// MI455X (gfx1250, wave32) 2-layer masked transformer.
// All matmuls on v_wmma_f32_16x16x32_f16 (f32->f16 inline convert, f32 acc).
// LDS tiles stored so the contraction dim is contiguous -> all fragment
// loads are 32-bit ds_load_b32. GEMM: 128x64 tile, 4 WMMA/wave/K-step.
// ---------------------------------------------------------------------------

typedef __attribute__((ext_vector_type(16))) _Float16     v16h;
typedef __attribute__((ext_vector_type(8)))  float        v8f;
typedef __attribute__((ext_vector_type(8)))  unsigned int v8u;
typedef __attribute__((ext_vector_type(4)))  unsigned int v4u;
typedef __attribute__((ext_vector_type(8)))  int          v8i;
typedef __attribute__((ext_vector_type(4)))  int          v4i;

#define NN   8192
#define DD   512
#define MM   512
#define HH   8
#define DK   64
#define LL   2
#define TT   2048
#define HID  1024

union HFrag { v8u u; v16h h; };

// Fragment load where the contraction dim (k) is CONTIGUOUS in memory:
// row index = lane&15 (A rows / B cols / keys), half = lane>>4.
// Per ISA 7.12.2: VGPR v holds k = (v>>2)*16 + (lane>>4)*8 + (v&3)*2, +1.
__device__ __forceinline__ v16h load_frag_k(const _Float16* base, int ld, int lane) {
    int m = lane & 15, hf = lane >> 4;
    HFrag f;
#pragma unroll
    for (int v = 0; v < 8; ++v) {
        int k = ((v >> 2) << 4) + (hf << 3) + ((v & 3) << 1);
        f.u[v] = *(const unsigned int*)(base + m * ld + k);
    }
    return f.h;
}

__device__ __forceinline__ v8f wmma16(v16h a, v16h b, v8f c) {
    return __builtin_amdgcn_wmma_f32_16x16x32_f16(false, a, false, b,
                                                  (short)0, c, false, false);
}

__device__ __forceinline__ float gelu_tanh(float x) {
    float x3 = x * x * x;
    return 0.5f * x * (1.0f + tanhf(0.7978845608028654f * (x + 0.044715f * x3)));
}

// ---------------- TDM warm: real tensor_load_to_lds + s_wait_tensorcnt -----
// Loads one 64x32 f32 tile of W_in into LDS via the Tensor Data Mover
// (warms L2; mainly exercises the CDNA5 async-tensor path). 1 wave.
// This toolchain exposes the 6-arg builtin:
//   (uint32x4 g0, int32x8 g1, int32x4 g2, int32x4 g3, int32x8 g4, i32 cpol)

__global__ void tdm_warm(const float* __restrict__ src, float* __restrict__ scratch) {
    __shared__ float buf[4096];
#if __has_builtin(__builtin_amdgcn_tensor_load_to_lds) && defined(__gfx1250__)
    unsigned long long ga = (unsigned long long)(size_t)src;
    v4u g0;
    g0[0] = 1u;                                    // count=1, user mode
    g0[1] = 0u;                                    // lds_addr = 0 (buf base)
    g0[2] = (unsigned int)(ga & 0xffffffffu);      // global_addr[31:0]
    g0[3] = (unsigned int)((ga >> 32) & 0x01ffffffu) | (2u << 30);  // [56:32] | type=2
    v8i g1;
    const unsigned int td0 = 64, td1 = 32, tile0 = 64, tile1 = 32, str0 = 64;
    g1[0] = (int)(2u << 16);                       // data_size=2 (4 bytes)
    g1[1] = (int)((td0 & 0xffffu) << 16);          // tensor_dim0[15:0] @ bit48
    g1[2] = (int)((td0 >> 16) | ((td1 & 0xffffu) << 16));
    g1[3] = (int)((td1 >> 16) | (tile0 << 16));    // tile_dim0 @ bit112
    g1[4] = (int)(tile1);                          // tile_dim1, tile_dim2=0
    g1[5] = (int)str0;                             // tensor_dim0_stride[31:0]
    g1[6] = 0;                                     // stride0[47:32], stride1[15:0]
    g1[7] = 0;
    v4i gz4 = {0, 0, 0, 0};
    v8i gz8 = {0, 0, 0, 0, 0, 0, 0, 0};
    __builtin_amdgcn_tensor_load_to_lds(g0, g1, gz4, gz4, gz8, 0);
#if __has_builtin(__builtin_amdgcn_s_wait_tensorcnt)
    __builtin_amdgcn_s_wait_tensorcnt(0);
#endif
    __syncthreads();
    scratch[threadIdx.x] = buf[threadIdx.x];       // keep the transfer live
#else
    buf[threadIdx.x] = src[threadIdx.x];
    __syncthreads();
    scratch[threadIdx.x] = buf[threadIdx.x];
#endif
}

// ---------------- stream compaction (valid_mask -> valid_idx, count) -------

__global__ void compact_kernel(const unsigned char* __restrict__ mask,
                               int* __restrict__ vidx, int* __restrict__ cnt) {
    __shared__ int ssum[1024];
    int tid = threadIdx.x;
    int base = tid * 8;
    int loc[8];
    int cl = 0;
#pragma unroll
    for (int j = 0; j < 8; ++j) {
        int v = (mask[base + j] != 0) ? 1 : 0;
        loc[j] = cl;
        cl += v;
    }
    ssum[tid] = cl;
    __syncthreads();
    for (int off = 1; off < 1024; off <<= 1) {
        int mine = ssum[tid];
        int add = (tid >= off) ? ssum[tid - off] : 0;
        __syncthreads();
        ssum[tid] = mine + add;
        __syncthreads();
    }
    int total = ssum[1023];
    int ex = ssum[tid] - cl;
#pragma unroll
    for (int j = 0; j < 8; ++j) {
        if (mask[base + j]) {
            int p = ex + loc[j];
            if (p < TT) vidx[p] = base + j;
        }
    }
    int c = total < TT ? total : TT;
    if (tid == 0) cnt[0] = c;
    __syncthreads();
    for (int p = c + tid; p < TT; p += 1024) vidx[p] = -1;
}

// ---------------- WMMA GEMM: C = epi(A@B + bias [,resid]) ------------------
// A: (Mr x K) f32, B: (K x Nc) f32, C: (Mr x Nc) f32.
// grid = (Mr/128, Nc/64), block 256 (8 waves); wave owns a 32x32 output.

__global__ void gemm_wmma(const float* __restrict__ A, const float* __restrict__ B,
                          const float* __restrict__ bias, const float* __restrict__ resid,
                          float* __restrict__ C, int Mr, int K, int Nc, int act) {
    __shared__ _Float16 As[128][40];   // rows x k (k contiguous)
    __shared__ _Float16 Bst[64][40];   // cols x k (k contiguous, transposed)

    int tid = threadIdx.x;
    int wave = tid >> 5, lane = tid & 31;
    int war = (wave >> 1) * 32;        // wave row offset: 0/32/64/96
    int wbc = (wave & 1) * 32;         // wave col offset: 0/32
    int bM = blockIdx.x * 128;
    int bN = blockIdx.y * 64;

    v8f acc00 = {}, acc01 = {}, acc10 = {}, acc11 = {};

    for (int k0 = 0; k0 < K; k0 += 32) {
        // stage A tile (128x32): 16 f32 per thread, contiguous f16 stores
        {
            int r = tid >> 1, c = (tid & 1) * 16;
            const float* src = A + (size_t)(bM + r) * K + k0 + c;
            float4 f0 = *(const float4*)(src);
            float4 f1 = *(const float4*)(src + 4);
            float4 f2 = *(const float4*)(src + 8);
            float4 f3 = *(const float4*)(src + 12);
            As[r][c + 0]  = (_Float16)f0.x; As[r][c + 1]  = (_Float16)f0.y;
            As[r][c + 2]  = (_Float16)f0.z; As[r][c + 3]  = (_Float16)f0.w;
            As[r][c + 4]  = (_Float16)f1.x; As[r][c + 5]  = (_Float16)f1.y;
            As[r][c + 6]  = (_Float16)f1.z; As[r][c + 7]  = (_Float16)f1.w;
            As[r][c + 8]  = (_Float16)f2.x; As[r][c + 9]  = (_Float16)f2.y;
            As[r][c + 10] = (_Float16)f2.z; As[r][c + 11] = (_Float16)f2.w;
            As[r][c + 12] = (_Float16)f3.x; As[r][c + 13] = (_Float16)f3.y;
            As[r][c + 14] = (_Float16)f3.z; As[r][c + 15] = (_Float16)f3.w;
        }
        // stage B tile (32x64) transposed -> Bst[col][k]
        {
            int r = tid >> 3, c = (tid & 7) * 8;   // r = k row, c = col base
            const float* src = B + (size_t)(k0 + r) * Nc + bN + c;
            float4 f0 = *(const float4*)(src);
            float4 f1 = *(const float4*)(src + 4);
            Bst[c + 0][r] = (_Float16)f0.x; Bst[c + 1][r] = (_Float16)f0.y;
            Bst[c + 2][r] = (_Float16)f0.z; Bst[c + 3][r] = (_Float16)f0.w;
            Bst[c + 4][r] = (_Float16)f1.x; Bst[c + 5][r] = (_Float16)f1.y;
            Bst[c + 6][r] = (_Float16)f1.z; Bst[c + 7][r] = (_Float16)f1.w;
            if (k0 + 32 < K)  // global_prefetch_b8 hint for next B tile
                __builtin_prefetch(B + (size_t)(k0 + 32 + r) * Nc + bN + c, 0, 1);
        }
        __syncthreads();

        v16h a0 = load_frag_k(&As[war][0], 40, lane);
        v16h a1 = load_frag_k(&As[war + 16][0], 40, lane);
        v16h b0 = load_frag_k(&Bst[wbc][0], 40, lane);
        v16h b1 = load_frag_k(&Bst[wbc + 16][0], 40, lane);
        acc00 = wmma16(a0, b0, acc00);
        acc01 = wmma16(a0, b1, acc01);
        acc10 = wmma16(a1, b0, acc10);
        acc11 = wmma16(a1, b1, acc11);
        __syncthreads();
    }

    // epilogue: C layout row m = v + 8*(lane>>4), col n = lane&15
    int n0 = bN + wbc + (lane & 15);
    int n1 = n0 + 16;
    int mr0 = bM + war + (lane >> 4) * 8;
    int mr1 = mr0 + 16;
    float bv0 = bias[n0], bv1 = bias[n1];
#pragma unroll
    for (int v = 0; v < 8; ++v) {
        float y00 = acc00[v] + bv0, y01 = acc01[v] + bv1;
        float y10 = acc10[v] + bv0, y11 = acc11[v] + bv1;
        if (act == 1) {
            y00 = gelu_tanh(y00); y01 = gelu_tanh(y01);
            y10 = gelu_tanh(y10); y11 = gelu_tanh(y11);
        }
        if (resid) {
            y00 += resid[(size_t)(mr0 + v) * Nc + n0];
            y01 += resid[(size_t)(mr0 + v) * Nc + n1];
            y10 += resid[(size_t)(mr1 + v) * Nc + n0];
            y11 += resid[(size_t)(mr1 + v) * Nc + n1];
        }
        C[(size_t)(mr0 + v) * Nc + n0] = y00;
        C[(size_t)(mr0 + v) * Nc + n1] = y01;
        C[(size_t)(mr1 + v) * Nc + n0] = y10;
        C[(size_t)(mr1 + v) * Nc + n1] = y11;
    }
}

// ---------------- tok gather: tok[t] = xp[idx]+e[idx]+f (or 0) -------------

__global__ void gather_tok(const float* __restrict__ xp, const float* __restrict__ e,
                           const float* __restrict__ f, const int* __restrict__ vidx,
                           float* __restrict__ tok) {
    int t = blockIdx.x, tid = threadIdx.x;
    int id = vidx[t];
    float4 o = make_float4(0.f, 0.f, 0.f, 0.f);
    if (id >= 0) {
        float4 a = ((const float4*)(xp + (size_t)id * MM))[tid];
        float4 b = ((const float4*)(e + (size_t)id * DD))[tid];
        float4 ff = ((const float4*)f)[tid];
        o.x = a.x + b.x + ff.x; o.y = a.y + b.y + ff.y;
        o.z = a.z + b.z + ff.z; o.w = a.w + b.w + ff.w;
    }
    ((float4*)(tok + (size_t)t * MM))[tid] = o;
}

// ---------------- RMSNorm over M=512 (optionally two scale sets) -----------

__global__ void rmsnorm_kernel(const float* __restrict__ tok,
                               const float* __restrict__ s1, float* __restrict__ o1,
                               const float* __restrict__ s2, float* __restrict__ o2) {
    __shared__ float red[4];
    int t = blockIdx.x, tid = threadIdx.x, lane = tid & 31;
    const float* row = tok + (size_t)t * MM;
    float4 x = ((const float4*)row)[tid];
    float ss = x.x * x.x + x.y * x.y + x.z * x.z + x.w * x.w;
#pragma unroll
    for (int m = 1; m < 32; m <<= 1) ss += __shfl_xor(ss, m, 32);
    if (lane == 0) red[tid >> 5] = ss;
    __syncthreads();
    float tot = red[0] + red[1] + red[2] + red[3];
    float inv = rsqrtf(tot * (1.0f / (float)MM) + 1e-6f);
    float4 g1 = ((const float4*)s1)[tid];
    float4 y;
    y.x = x.x * inv * (1.f + g1.x); y.y = x.y * inv * (1.f + g1.y);
    y.z = x.z * inv * (1.f + g1.z); y.w = x.w * inv * (1.f + g1.w);
    ((float4*)(o1 + (size_t)t * MM))[tid] = y;
    if (o2) {
        float4 g2 = ((const float4*)s2)[tid];
        y.x = x.x * inv * (1.f + g2.x); y.y = x.y * inv * (1.f + g2.y);
        y.z = x.z * inv * (1.f + g2.z); y.w = x.w * inv * (1.f + g2.w);
        ((float4*)(o2 + (size_t)t * MM))[tid] = y;
    }
}

// ---------------- per-head RMS + RoPE, f32 (T,H*64) -> f16 (H,T,64) --------

__global__ void qk_prep(const float* __restrict__ In, const float* __restrict__ scale,
                        _Float16* __restrict__ Out, int do_rope) {
    int t = blockIdx.x;
    int head = threadIdx.x >> 5, lane = threadIdx.x & 31;
    const float* row = In + (size_t)t * (HH * DK) + head * DK;
    float a = row[lane], b = row[lane + 32];
    float ss = a * a + b * b;
#pragma unroll
    for (int m = 1; m < 32; m <<= 1) ss += __shfl_xor(ss, m, 32);
    float inv = rsqrtf(ss * (1.0f / (float)DK) + 1e-6f);
    float ga = a * inv * (1.f + scale[lane]);
    float gb = b * inv * (1.f + scale[lane + 32]);
    float oa = ga, ob = gb;
    if (do_rope) {
        float frac = (2.0f * (float)lane) / (float)DK;
        float ts = __powf(10000.0f, frac);
        float si = (float)t / ts;
        float s, c;
        __sincosf(si, &s, &c);
        oa = ga * c - gb * s;
        ob = gb * c + ga * s;
    }
    _Float16* orow = Out + ((size_t)head * TT + t) * DK;
    orow[lane] = (_Float16)oa;
    orow[lane + 32] = (_Float16)ob;
}

__global__ void v_prep(const float* __restrict__ In, _Float16* __restrict__ Out) {
    int i = blockIdx.x * blockDim.x + threadIdx.x;
    int d = i & (DK - 1);
    int h = (i >> 6) & (HH - 1);
    int t = i >> 9;
    Out[((size_t)h * TT + t) * DK + d] = (_Float16)In[i];
}

// ---------------- flash attention: one head x 64 query rows per block ------
// grid = (T/64, H), block 128 (4 waves, 16 query rows each).

__global__ void attn_flash(const _Float16* __restrict__ Q, const _Float16* __restrict__ K,
                           const _Float16* __restrict__ V, float* __restrict__ Out,
                           const float* __restrict__ head_scale, const int* __restrict__ cntp) {
    __shared__ _Float16 Ks[32][80];    // [key][feature] (feature contiguous)
    __shared__ _Float16 Vst[64][40];   // [feature][key] (key contiguous)
    __shared__ _Float16 Ps[4][16][40]; // per-wave P tile [row][key]

    int head = blockIdx.y;
    int tid = threadIdx.x, wave = tid >> 5, lane = tid & 31;
    int q0 = blockIdx.x * 64 + wave * 16;
    int nvalid = cntp[0];

    const _Float16* Qh = Q + (size_t)head * TT * DK;
    const _Float16* Kh = K + (size_t)head * TT * DK;
    const _Float16* Vh = V + (size_t)head * TT * DK;

    // Q tile A-fragments (16x64 as two K=32 fragments), loaded once from global
    v16h qa0 = load_frag_k(Qh + (size_t)q0 * DK, DK, lane);
    v16h qa1 = load_frag_k(Qh + (size_t)q0 * DK + 32, DK, lane);

    v8f o0 = {}, o1 = {}, o2 = {}, o3 = {};
    float rm[8], rl[8];
#pragma unroll
    for (int v = 0; v < 8; ++v) { rm[v] = -3e38f; rl[v] = 0.f; }

    for (int kc = 0; kc < TT; kc += 32) {
        // stage K (row-major) and V (transposed) chunks
        {
            int r = tid >> 3, c = (tid & 7) * 8;
#pragma unroll
            for (int rep = 0; rep < 2; ++rep) {
                int rr = r + rep * 16;
                *(float4*)&Ks[rr][c] = *(const float4*)&Kh[(size_t)(kc + rr) * DK + c];
                const _Float16* vr = &Vh[(size_t)(kc + rr) * DK + c];
#pragma unroll
                for (int j = 0; j < 8; ++j) Vst[c + j][rr] = vr[j];
            }
        }
        __syncthreads();

        // S(16x32) = Q . K^T : two 16x16 subtiles, 2 WMMA each (DK=2x32)
        v8f s0 = {}, s1 = {};
        s0 = wmma16(qa0, load_frag_k(&Ks[0][0], 80, lane), s0);
        s0 = wmma16(qa1, load_frag_k(&Ks[0][32], 80, lane), s0);
        s1 = wmma16(qa0, load_frag_k(&Ks[16][0], 80, lane), s1);
        s1 = wmma16(qa1, load_frag_k(&Ks[16][32], 80, lane), s1);

        const float sc = 0.125f;  // 1/sqrt(64)
        int key0 = kc + (lane & 15);
        int key1 = key0 + 16;
#pragma unroll
        for (int v = 0; v < 8; ++v) {
            s0[v] = (key0 < nvalid) ? s0[v] * sc : -1e30f;
            s1[v] = (key1 < nvalid) ? s1[v] * sc : -1e30f;
        }
        float nm[8];
#pragma unroll
        for (int v = 0; v < 8; ++v) nm[v] = fmaxf(s0[v], s1[v]);
#pragma unroll
        for (int m = 1; m < 16; m <<= 1)
#pragma unroll
            for (int v = 0; v < 8; ++v) nm[v] = fmaxf(nm[v], __shfl_xor(nm[v], m, 32));
        float resc[8];
#pragma unroll
        for (int v = 0; v < 8; ++v) {
            float mnew = fmaxf(rm[v], nm[v]);
            resc[v] = __expf(rm[v] - mnew);
            rm[v] = mnew;
        }
        float p0[8], p1[8], rs[8];
#pragma unroll
        for (int v = 0; v < 8; ++v) {
            p0[v] = __expf(s0[v] - rm[v]);
            p1[v] = __expf(s1[v] - rm[v]);
            rs[v] = p0[v] + p1[v];
        }
#pragma unroll
        for (int m = 1; m < 16; m <<= 1)
#pragma unroll
            for (int v = 0; v < 8; ++v) rs[v] += __shfl_xor(rs[v], m, 32);
#pragma unroll
        for (int v = 0; v < 8; ++v) {
            rl[v] = rl[v] * resc[v] + rs[v];
            o0[v] *= resc[v]; o1[v] *= resc[v];
            o2[v] *= resc[v]; o3[v] *= resc[v];
        }
        // stage P (16x32 f16) per-wave; same-wave LDS ordering is in-order
        {
            int n = lane & 15, hf = lane >> 4;
#pragma unroll
            for (int v = 0; v < 8; ++v) {
                Ps[wave][v + 8 * hf][n]      = (_Float16)p0[v];
                Ps[wave][v + 8 * hf][n + 16] = (_Float16)p1[v];
            }
        }
        v16h pa = load_frag_k(&Ps[wave][0][0], 40, lane);
        // O(16x64) += P(16x32) . V(32x64): 4 WMMAs (K=32 exactly)
        o0 = wmma16(pa, load_frag_k(&Vst[0][0], 40, lane), o0);
        o1 = wmma16(pa, load_frag_k(&Vst[16][0], 40, lane), o1);
        o2 = wmma16(pa, load_frag_k(&Vst[32][0], 40, lane), o2);
        o3 = wmma16(pa, load_frag_k(&Vst[48][0], 40, lane), o3);
        __syncthreads();
    }

    float hsc = 1.0f + head_scale[head];
    int n = lane & 15, hf = lane >> 4;
#pragma unroll
    for (int v = 0; v < 8; ++v) {
        int t = q0 + v + 8 * hf;
        float inv = hsc / fmaxf(rl[v], 1e-20f);
        float* orow = Out + (size_t)t * (HH * DK) + head * DK;
        orow[n]      = o0[v] * inv;
        orow[n + 16] = o1[v] * inv;
        orow[n + 32] = o2[v] * inv;
        orow[n + 48] = o3[v] * inv;
    }
}

// ---------------- pos_mask row zeroing, output zero + scatter --------------

__global__ void mask_rows(float* __restrict__ tok, const int* __restrict__ cntp) {
    int t = blockIdx.x;
    if (t < cntp[0]) return;
    ((float4*)(tok + (size_t)t * MM))[threadIdx.x] = make_float4(0.f, 0.f, 0.f, 0.f);
}

__global__ void zero_out(float* __restrict__ out) {
    size_t i = (size_t)blockIdx.x * blockDim.x + threadIdx.x;
    ((float4*)out)[i] = make_float4(0.f, 0.f, 0.f, 0.f);
}

__global__ void scatter_out(const float* __restrict__ tok, const int* __restrict__ vidx,
                            const int* __restrict__ cntp, float* __restrict__ out) {
    int t = blockIdx.x;
    if (t >= cntp[0]) return;
    int id = vidx[t];
    if (id < 0) return;
    ((float4*)(out + (size_t)id * MM))[threadIdx.x] =
        ((const float4*)(tok + (size_t)t * MM))[threadIdx.x];
}

// ---------------------------------------------------------------------------

extern "C" void kernel_launch(void* const* d_in, const int* in_sizes, int n_in,
                              void* d_out, int out_size, void* d_ws, size_t ws_size,
                              hipStream_t stream) {
    const float* x     = (const float*)d_in[0];
    const float* e     = (const float*)d_in[1];
    const float* f     = (const float*)d_in[2];
    const float* W_in  = (const float*)d_in[3];
    const float* b_in  = (const float*)d_in[4];
    const float* ln_q  = (const float*)d_in[5];
    const float* ln_kv = (const float*)d_in[6];
    const float* Wq    = (const float*)d_in[7];
    const float* bq    = (const float*)d_in[8];
    const float* Wk    = (const float*)d_in[9];
    const float* bk    = (const float*)d_in[10];
    const float* Wv    = (const float*)d_in[11];
    const float* bv    = (const float*)d_in[12];
    const float* qln   = (const float*)d_in[13];
    const float* kln   = (const float*)d_in[14];
    const float* hsc   = (const float*)d_in[15];
    const float* Wo    = (const float*)d_in[16];
    const float* bo    = (const float*)d_in[17];
    const float* lnf   = (const float*)d_in[18];
    const float* W1    = (const float*)d_in[19];
    const float* b1    = (const float*)d_in[20];
    const float* W2    = (const float*)d_in[21];
    const float* b2    = (const float*)d_in[22];
    const unsigned char* vmask = (const unsigned char*)d_in[23];  // numpy bool

    char* w = (char*)d_ws;
    size_t off = 0;
    int*   cnt  = (int*)(w + off);
    float* tdms = (float*)(w + 64);             // dead scratch for tdm_warm
    off = 256;
    int* vidx = (int*)(w + off); off += (size_t)TT * 4; off = (off + 16383) & ~(size_t)16383;
    float* xp   = (float*)(w + off); off += (size_t)NN * MM * 4;
    float* tok  = (float*)(w + off); off += (size_t)TT * MM * 4;
    float* qn   = (float*)(w + off); off += (size_t)TT * MM * 4;
    float* kvn  = (float*)(w + off); off += (size_t)TT * MM * 4;
    float* qh   = (float*)(w + off); off += (size_t)TT * MM * 4;
    float* kh   = (float*)(w + off); off += (size_t)TT * MM * 4;
    float* vh   = (float*)(w + off); off += (size_t)TT * MM * 4;
    _Float16* Qh = (_Float16*)(w + off); off += (size_t)TT * MM * 2;
    _Float16* Kh = (_Float16*)(w + off); off += (size_t)TT * MM * 2;
    _Float16* Vh = (_Float16*)(w + off); off += (size_t)TT * MM * 2;
    float* attn = (float*)(w + off); off += (size_t)TT * MM * 4;
    float* hln  = (float*)(w + off); off += (size_t)TT * MM * 4;
    float* hh   = (float*)(w + off); off += (size_t)TT * HID * 4;
    (void)ws_size; (void)in_sizes; (void)n_in; (void)out_size;

    tdm_warm<<<1, 32, 0, stream>>>(W_in, tdms);
    compact_kernel<<<1, 1024, 0, stream>>>(vmask, vidx, cnt);

    // xp = x @ W_in + b_in   (8192x512x512)
    gemm_wmma<<<dim3(NN / 128, MM / 64), 256, 0, stream>>>(x, W_in, b_in, nullptr, xp,
                                                           NN, DD, MM, 0);
    gather_tok<<<TT, 128, 0, stream>>>(xp, e, f, vidx, tok);

    for (int i = 0; i < LL; ++i) {
        const float* Wqi = Wq + (size_t)i * MM * HH * DK;
        const float* Wki = Wk + (size_t)i * MM * HH * DK;
        const float* Wvi = Wv + (size_t)i * MM * HH * DK;
        const float* Woi = Wo + (size_t)i * HH * DK * MM;
        const float* W1i = W1 + (size_t)i * MM * HID;
        const float* W2i = W2 + (size_t)i * HID * MM;

        rmsnorm_kernel<<<TT, 128, 0, stream>>>(tok, ln_q + (size_t)i * MM, qn,
                                               ln_kv + (size_t)i * MM, kvn);
        gemm_wmma<<<dim3(TT / 128, MM / 64), 256, 0, stream>>>(qn, Wqi, bq + (size_t)i * MM,
                                                               nullptr, qh, TT, MM, MM, 0);
        gemm_wmma<<<dim3(TT / 128, MM / 64), 256, 0, stream>>>(kvn, Wki, bk + (size_t)i * MM,
                                                               nullptr, kh, TT, MM, MM, 0);
        gemm_wmma<<<dim3(TT / 128, MM / 64), 256, 0, stream>>>(kvn, Wvi, bv + (size_t)i * MM,
                                                               nullptr, vh, TT, MM, MM, 0);
        qk_prep<<<TT, 256, 0, stream>>>(qh, qln + (size_t)i * DK, Qh, 1);
        qk_prep<<<TT, 256, 0, stream>>>(kh, kln + (size_t)i * DK, Kh, 1);
        v_prep<<<(TT * HH * DK) / 256, 256, 0, stream>>>(vh, Vh);
        attn_flash<<<dim3(TT / 64, HH), 128, 0, stream>>>(Qh, Kh, Vh, attn,
                                                          hsc + (size_t)i * HH, cnt);
        gemm_wmma<<<dim3(TT / 128, MM / 64), 256, 0, stream>>>(attn, Woi, bo + (size_t)i * MM,
                                                               tok, tok, TT, MM, MM, 0);
        rmsnorm_kernel<<<TT, 128, 0, stream>>>(tok, lnf + (size_t)i * MM, hln,
                                               nullptr, nullptr);
        gemm_wmma<<<dim3(TT / 128, HID / 64), 256, 0, stream>>>(hln, W1i, b1 + (size_t)i * HID,
                                                                nullptr, hh, TT, MM, HID, 1);
        gemm_wmma<<<dim3(TT / 128, MM / 64), 256, 0, stream>>>(hh, W2i, b2 + (size_t)i * MM,
                                                               tok, tok, TT, HID, MM, 0);
        mask_rows<<<TT, 128, 0, stream>>>(tok, cnt);
    }

    zero_out<<<(NN * MM / 4) / 256, 256, 0, stream>>>((float*)d_out);
    scatter_out<<<TT, 128, 0, stream>>>(tok, vidx, cnt, (float*)d_out);
}
